// CTGRU_28381143891986
// MI455X (gfx1250) — matrix operand order repaired and checked by reading, not compiled, not run on hardware
//
#include <hip/hip_runtime.h>
#include <hip/hip_bf16.h>

typedef __attribute__((ext_vector_type(16))) __bf16 v16bf;
typedef __attribute__((ext_vector_type(8)))  float  v8f;
typedef unsigned short u16;
typedef unsigned int   u32;

#define BATCH 2048
#define UNITS 512
#define MB    8
#define DIN   512
#define FDIM  1024   // DIN + UNITS
#define NR    4096   // UNITS * MB
#define ND    512

// 0.5 * ln(10)
#define HLN10 1.1512925464970229f

struct U16x16 { uint4 a, b; };   // 32 bytes == v16bf

__device__ __forceinline__ u16 f2bf(float f) {
    u32 u = __builtin_bit_cast(u32, f);
    u32 r = u + 0x7FFFu + ((u >> 16) & 1u);   // round-to-nearest-even
    return (u16)(r >> 16);
}

// gfx1250 async global->LDS copy (16B per lane), tracked by ASYNCcnt
__device__ __forceinline__ void async_ld_b128(u32 lds_byte_addr, const void* gaddr) {
    asm volatile("global_load_async_to_lds_b128 %0, %1, off"
                 :: "v"(lds_byte_addr), "v"(gaddr) : "memory");
}
__device__ __forceinline__ void wait_async0() {
    asm volatile("s_wait_asynccnt 0x0" ::: "memory");
}

// ---------------------------------------------------------------- conversions
// inputs[b][0..511] -> dst[b][0..511] (bf16, row stride FDIM)
__global__ void k_pack_inputs(const float* __restrict__ in,
                              u16* __restrict__ dst) {
    int i = blockIdx.x * blockDim.x + threadIdx.x;     // BATCH * DIN / 4
    if (i >= BATCH * (DIN / 4)) return;
    int b  = i / (DIN / 4);
    int d4 = i % (DIN / 4);
    float4 v = ((const float4*)in)[(size_t)b * (DIN / 4) + d4];
    ushort4 o;
    o.x = f2bf(v.x); o.y = f2bf(v.y); o.z = f2bf(v.z); o.w = f2bf(v.w);
    ((ushort4*)(dst + (size_t)b * FDIM))[d4] = o;
}

// h[b][u] = sum_M state -> fused[b][DIN+u]
__global__ void k_pack_h(const float* __restrict__ state,
                         u16* __restrict__ fused) {
    int i = blockIdx.x * blockDim.x + threadIdx.x;     // BATCH * UNITS
    if (i >= BATCH * UNITS) return;
    int b = i >> 9, u = i & 511;
    const float4* p = (const float4*)(state + (size_t)b * NR + u * MB);
    float4 a = p[0], c = p[1];
    float s = a.x + a.y + a.z + a.w + c.x + c.y + c.z + c.w;
    fused[(size_t)b * FDIM + DIN + u] = f2bf(s);
}

// W[k][n] (f32, row-major K x N) -> Wt[n][k] (bf16, N x FDIM)
__global__ __launch_bounds__(256)
void k_transpose_bf16(const float* __restrict__ W, u16* __restrict__ Wt, int N) {
    __shared__ float tile[32][33];
    int n0 = blockIdx.x * 32, k0 = blockIdx.y * 32;
    int tx = threadIdx.x & 31, ty = threadIdx.x >> 5;  // 32 x 8
#pragma unroll
    for (int i = 0; i < 32; i += 8)
        tile[ty + i][tx] = W[(size_t)(k0 + ty + i) * N + n0 + tx];
    __syncthreads();
#pragma unroll
    for (int i = 0; i < 32; i += 8)
        Wt[(size_t)(n0 + ty + i) * FDIM + k0 + tx] = f2bf(tile[tx][ty + i]);
}

// ---------------------------------------------------------------- WMMA GEMM
// C[2048 x N] = A[2048 x 1024](bf16) @ Bt^T (Bt is [N x 1024] bf16) + bias[N]
__global__ __launch_bounds__(256)
void k_gemm(const u16* __restrict__ A, const u16* __restrict__ Bt,
            const float* __restrict__ bias, float* __restrict__ C, int N) {
    constexpr int LST = 40;                       // padded LDS stride (halves)
    constexpr int TSZ = 128 * LST;
    __shared__ alignas(16) u16 As[2][TSZ];
    __shared__ alignas(16) u16 Bs[2][TSZ];

    const int t    = threadIdx.x;
    const int m0   = blockIdx.y * 128;
    const int n0   = blockIdx.x * 128;
    const int w    = t >> 5, lane = t & 31;
    const int wm   = w >> 2, wn = w & 3;          // 2 x 4 wave grid
    const int lh   = lane >> 4, lr = lane & 15;   // half-wave select / sub-lane

    const u32 aBase0 = (u32)(uintptr_t)&As[0][0];
    const u32 aBase1 = (u32)(uintptr_t)&As[1][0];
    const u32 bBase0 = (u32)(uintptr_t)&Bs[0][0];
    const u32 bBase1 = (u32)(uintptr_t)&Bs[1][0];

    v8f acc[4][2];
#pragma unroll
    for (int i = 0; i < 4; i++)
#pragma unroll
        for (int j = 0; j < 2; j++)
#pragma unroll
            for (int r = 0; r < 8; r++) acc[i][j][r] = 0.f;

    // issue async copies for K-chunk kk into buffer p (4 x b128 per thread)
    auto issue = [&](int kk, int p) {
        const int k0 = kk * 32;
        const u32 ab = p ? aBase1 : aBase0;
        const u32 bb = p ? bBase1 : bBase0;
#pragma unroll
        for (int uu = 0; uu < 2; uu++) {
            int id  = t + uu * 256;               // 0..511
            int row = id >> 2, ch = id & 3;
            u32 loff = (u32)(row * LST + ch * 8) * 2;
            async_ld_b128(ab + loff, A  + (size_t)(m0 + row) * FDIM + k0 + ch * 8);
            async_ld_b128(bb + loff, Bt + (size_t)(n0 + row) * FDIM + k0 + ch * 8);
        }
    };

    issue(0, 0);
    for (int kk = 0; kk < FDIM / 32; kk++) {
        const int p = kk & 1;
        wait_async0();            // my tile-kk copies done
        __syncthreads();          // everyone's copies done
        if (kk + 1 < FDIM / 32) issue(kk + 1, p ^ 1);

        const u16* as = &As[p][0];
        const u16* bs = &Bs[p][0];

        // A fragment: lane<16 holds K0..7 + K16..23 of row M=lr (ISA layout)
        v16bf af[4], bfr[2];
#pragma unroll
        for (int mt = 0; mt < 4; mt++) {
            int row = wm * 64 + mt * 16 + lr;
            const u16* pp = as + row * LST + lh * 8;
            U16x16 tmp;
            tmp.a = *(const uint4*)pp;            // V0-3: K = lh*8 .. +7
            tmp.b = *(const uint4*)(pp + 16);     // V4-7: K = lh*8+16 .. +7
            af[mt] = __builtin_bit_cast(v16bf, tmp);
        }
        // B fragment: lane<16 holds K0..7 + K16..23 of column N=lr
#pragma unroll
        for (int nt = 0; nt < 2; nt++) {
            int col = wn * 32 + nt * 16 + lr;
            const u16* pp = bs + col * LST + lh * 8;
            U16x16 tmp;
            tmp.a = *(const uint4*)pp;
            tmp.b = *(const uint4*)(pp + 16);
            bfr[nt] = __builtin_bit_cast(v16bf, tmp);
        }
#pragma unroll
        for (int mt = 0; mt < 4; mt++)
#pragma unroll
            for (int nt = 0; nt < 2; nt++)
                acc[mt][nt] = __builtin_amdgcn_wmma_f32_16x16x32_bf16(
                    false, af[mt], false, bfr[nt],
                    (short)0, acc[mt][nt], false, false);
        __syncthreads();          // done reading buf p before it is rewritten
    }

    // epilogue: C/D layout -> VGPR r holds rows r (lanes 0-15) / r+8 (16-31)
#pragma unroll
    for (int nt = 0; nt < 2; nt++) {
        int col = n0 + wn * 32 + nt * 16 + lr;
        float bv = bias[col];
#pragma unroll
        for (int mt = 0; mt < 4; mt++) {
            int rowb = m0 + wm * 64 + mt * 16 + lh * 8;
#pragma unroll
            for (int r = 0; r < 8; r++)
                C[(size_t)(rowb + r) * N + col] = acc[mt][nt][r] + bv;
        }
    }
}

// ---------------------------------------------------------------- gates
// rki softmax + q_input -> reset[b][DIN+u] (bf16)
__global__ void k_gate(const float* __restrict__ lnr,
                       const float* __restrict__ state,
                       u16* __restrict__ resetb) {
    int i = blockIdx.x * blockDim.x + threadIdx.x;     // BATCH * UNITS
    if (i >= BATCH * UNITS) return;
    int b = i >> 9, u = i & 511;
    const float4* xp = (const float4*)(lnr   + (size_t)b * NR + u * MB);
    const float4* hp = (const float4*)(state + (size_t)b * NR + u * MB);
    float4 x01 = xp[0], x23 = xp[1];
    float4 h01 = hp[0], h23 = hp[1];
    float x[MB] = {x01.x, x01.y, x01.z, x01.w, x23.x, x23.y, x23.z, x23.w};
    float hh[MB] = {h01.x, h01.y, h01.z, h01.w, h23.x, h23.y, h23.z, h23.w};
    float e[MB], mx = -3.4e38f;
#pragma unroll
    for (int j = 0; j < MB; j++) {
        float d = x[j] - HLN10 * (float)j;
        e[j] = -d * d;
        mx = fmaxf(mx, e[j]);
    }
    float s = 0.f, q = 0.f;
#pragma unroll
    for (int j = 0; j < MB; j++) {
        float wgt = __expf(e[j] - mx);
        s += wgt;
        q += wgt * hh[j];
    }
    resetb[(size_t)b * FDIM + DIN + u] = f2bf(q / s);
}

// ski softmax + tanh candidate + decay -> h_next, h_hat_next
__global__ void k_final(const float* __restrict__ lns,
                        const float* __restrict__ state,
                        const float* __restrict__ dlin,
                        const float* __restrict__ elapsed,
                        float* __restrict__ out) {
    int i = blockIdx.x * blockDim.x + threadIdx.x;     // BATCH * UNITS
    if (i >= BATCH * UNITS) return;
    int b = i >> 9, u = i & 511;
    float* out_h  = out;                               // [B, U]
    float* out_hh = out + (size_t)BATCH * UNITS;       // [B, U*M]
    const float4* xp = (const float4*)(lns   + (size_t)b * NR + u * MB);
    const float4* hp = (const float4*)(state + (size_t)b * NR + u * MB);
    float4 x01 = xp[0], x23 = xp[1];
    float4 h01 = hp[0], h23 = hp[1];
    float x[MB] = {x01.x, x01.y, x01.z, x01.w, x23.x, x23.y, x23.z, x23.w};
    float hh[MB] = {h01.x, h01.y, h01.z, h01.w, h23.x, h23.y, h23.z, h23.w};
    float qk = tanhf(dlin[(size_t)b * UNITS + u]);
    float el = elapsed[b];

    float e[MB], mx = -3.4e38f;
#pragma unroll
    for (int j = 0; j < MB; j++) {
        float d = x[j] - HLN10 * (float)j;
        e[j] = -d * d;
        mx = fmaxf(mx, e[j]);
    }
    float wgt[MB], s = 0.f;
#pragma unroll
    for (int j = 0; j < MB; j++) { wgt[j] = __expf(e[j] - mx); s += wgt[j]; }
    float inv = 1.f / s;
    float hn[MB], hsum = 0.f;
#pragma unroll
    for (int j = 0; j < MB; j++) {
        float sk = wgt[j] * inv;
        float et = __expf(-el * __expf(-HLN10 * (float)j));  // exp(-elapsed/tau_j)
        hn[j] = ((1.f - sk) * hh[j] + sk * qk) * et;
        hsum += hn[j];
    }
    float4* op = (float4*)(out_hh + (size_t)b * NR + u * MB);
    op[0] = make_float4(hn[0], hn[1], hn[2], hn[3]);
    op[1] = make_float4(hn[4], hn[5], hn[6], hn[7]);
    out_h[(size_t)b * UNITS + u] = hsum;
}

// ---------------------------------------------------------------- launcher
extern "C" void kernel_launch(void* const* d_in, const int* in_sizes, int n_in,
                              void* d_out, int out_size, void* d_ws, size_t ws_size,
                              hipStream_t stream) {
    const float* inputs  = (const float*)d_in[0];
    const float* elapsed = (const float*)d_in[1];
    const float* state   = (const float*)d_in[2];
    const float* W_r     = (const float*)d_in[3];
    const float* b_r     = (const float*)d_in[4];
    const float* W_d     = (const float*)d_in[5];
    const float* b_d     = (const float*)d_in[6];
    const float* W_u     = (const float*)d_in[7];
    const float* b_u     = (const float*)d_in[8];
    (void)in_sizes; (void)n_in; (void)out_size; (void)ws_size;

    char* ws = (char*)d_ws;
    size_t off = 0;
    auto take = [&](size_t bytes) {
        void* p = ws + off;
        off += (bytes + 255) & ~(size_t)255;
        return p;
    };
    u16*   fused  = (u16*)take((size_t)BATCH * FDIM * 2);
    u16*   resetb = (u16*)take((size_t)BATCH * FDIM * 2);
    u16*   Wrt    = (u16*)take((size_t)NR * FDIM * 2);   // W_r^T [N][K]
    u16*   Wut    = (u16*)take((size_t)NR * FDIM * 2);   // W_u^T
    u16*   Wdt    = (u16*)take((size_t)ND * FDIM * 2);   // W_d^T
    float* lnr    = (float*)take((size_t)BATCH * NR * 4);
    float* lns    = (float*)take((size_t)BATCH * NR * 4);
    float* dlin   = (float*)take((size_t)BATCH * ND * 4);

    const int T = 256;
    // transpose + convert weights to bf16 [N][K]
    k_transpose_bf16<<<dim3(NR / 32, FDIM / 32), T, 0, stream>>>(W_r, Wrt, NR);
    k_transpose_bf16<<<dim3(NR / 32, FDIM / 32), T, 0, stream>>>(W_u, Wut, NR);
    k_transpose_bf16<<<dim3(ND / 32, FDIM / 32), T, 0, stream>>>(W_d, Wdt, ND);
    // build fused / reset activations (bf16)
    k_pack_inputs<<<(BATCH * DIN / 4 + T - 1) / T, T, 0, stream>>>(inputs, fused);
    k_pack_inputs<<<(BATCH * DIN / 4 + T - 1) / T, T, 0, stream>>>(inputs, resetb);
    k_pack_h<<<(BATCH * UNITS + T - 1) / T, T, 0, stream>>>(state, fused);

    // ln_tau_r = fused @ W_r + b_r ; ln_tau_s = fused @ W_u + b_u
    k_gemm<<<dim3(NR / 128, BATCH / 128), dim3(T), 0, stream>>>(fused, Wrt, b_r, lnr, NR);
    k_gemm<<<dim3(NR / 128, BATCH / 128), dim3(T), 0, stream>>>(fused, Wut, b_u, lns, NR);

    // retrieval gate -> reset_value
    k_gate<<<(BATCH * UNITS + T - 1) / T, T, 0, stream>>>(lnr, state, resetb);

    // d_lin = reset_value @ W_d + b_d
    k_gemm<<<dim3(ND / 128, BATCH / 128), dim3(T), 0, stream>>>(resetb, Wdt, b_d, dlin, ND);

    // final gates, decay, outputs
    k_final<<<(BATCH * UNITS + T - 1) / T, T, 0, stream>>>(lns, state, dlin, elapsed,
                                                           (float*)d_out);
}
